// DeformGNN_86517821214969
// MI455X (gfx1250) — matrix operand order repaired and checked by reading, not compiled, not run on hardware
//
#include <hip/hip_runtime.h>

typedef __attribute__((ext_vector_type(16))) _Float16 v16h;
typedef __attribute__((ext_vector_type(8)))  float    v8f;

#define NPTS 1024
#define SOUT 256
#define FDIM 256            // uniform K dimension for every gconv layer
#define FP   264            // padded LDS row stride (16B aligned)

union AB16 { v16h h; uint4 u[2]; };

static __device__ inline unsigned pk2(float a, float b)
{
    union { _Float16 h[2]; unsigned u; } t;
    t.h[0] = (_Float16)a;
    t.h[1] = (_Float16)b;
    return t.u;
}

// ---------------------------------------------------------------------------
// Bilinear interpolation -> xbuf f32 (B,N,256) and xt f16 (B,256,N)
// cols 0..127 features, 128..129 base_point, 130..255 zero padding.
// grid (N, B), block 256
// ---------------------------------------------------------------------------
__global__ __launch_bounds__(256) void interp_kernel(
    const float* __restrict__ feat,   // B,128,128,128
    const float* __restrict__ bp,     // B,N,2
    float* __restrict__ xbuf,         // B,N,256
    _Float16* __restrict__ xt)        // B,256,N
{
    const int n = blockIdx.x;
    const int b = blockIdx.y;
    const int c = threadIdx.x;

    const float bx = bp[((size_t)b * NPTS + n) * 2 + 0];
    const float by = bp[((size_t)b * NPTS + n) * 2 + 1];
    const float Xs = bx * 128.f;
    const float Ys = by * 128.f;
    const float X0f = floorf(Xs), Y0f = floorf(Ys);
    const float X1f = X0f + 1.f,  Y1f = Y0f + 1.f;
    const float w00 = (X1f - Xs) * (Y1f - Ys);
    const float w01 = (X1f - Xs) * (Ys - Y0f);
    const float w10 = (Xs - X0f) * (Y1f - Ys);
    const float w11 = (Xs - X0f) * (Ys - Y0f);
    int x0 = (int)X0f; x0 = x0 < 0 ? 0 : (x0 > 127 ? 127 : x0);
    int x1 = (int)X1f; x1 = x1 < 0 ? 0 : (x1 > 127 ? 127 : x1);
    int y0 = (int)Y0f; y0 = y0 < 0 ? 0 : (y0 > 127 ? 127 : y0);
    int y1 = (int)Y1f; y1 = y1 < 0 ? 0 : (y1 > 127 ? 127 : y1);

    float v;
    if (c < 128) {
        const float* fb = feat + ((size_t)b * 128 + c) * 128 * 128;
        v = w00 * fb[y0 * 128 + x0] + w01 * fb[y1 * 128 + x0] +
            w10 * fb[y0 * 128 + x1] + w11 * fb[y1 * 128 + x1];
    } else if (c == 128) {
        v = bx;
    } else if (c == 129) {
        v = by;
    } else {
        v = 0.f;   // K padding 130..255
    }
    xbuf[((size_t)b * NPTS + n) * FDIM + c] = v;
    xt[((size_t)b * FDIM + c) * NPTS + n]   = (_Float16)v;
}

// ---------------------------------------------------------------------------
// Generic f32 -> f16 conversion (n multiple of 4)
// ---------------------------------------------------------------------------
__global__ void cvt_h_kernel(const float* __restrict__ s,
                             _Float16* __restrict__ d, int n)
{
    int i = (blockIdx.x * blockDim.x + threadIdx.x) * 4;
    if (i >= n) return;
    float4 v = *(const float4*)(s + i);
    d[i + 0] = (_Float16)v.x;
    d[i + 1] = (_Float16)v.y;
    d[i + 2] = (_Float16)v.z;
    d[i + 3] = (_Float16)v.w;
}

// ---------------------------------------------------------------------------
// first_W (2,256,130) f32 -> (2,256,256) f16 zero-padded
// ---------------------------------------------------------------------------
__global__ void padw16_kernel(const float* __restrict__ W,
                              _Float16* __restrict__ Wp)
{
    int i = blockIdx.x * blockDim.x + threadIdx.x;
    if (i >= 2 * 256 * FDIM) return;
    int k = i % FDIM;
    int sr = i / FDIM;
    Wp[i] = (k < 130) ? (_Float16)W[sr * 130 + k] : (_Float16)0.f;
}

// ---------------------------------------------------------------------------
// Fused graph-conv, uniform F=256, 32-row macro tile:
//   Y = act( X @ W0^T + b0 + (adj @ X) @ W1^T + b1 [+ Res] )
// grid (N/32, B), block 128 (4 waves). Wave w owns 32 rows x 64 columns:
// every streamed B fragment (Xt column / weight fragment) feeds TWO A
// fragments, halving loads per WMMA (ratio 3 -> 1.5).
// ---------------------------------------------------------------------------
__global__ __launch_bounds__(128) void gconv_kernel(
    const float*    __restrict__ X,     // B,N,256 f32 (row tile source)
    const _Float16* __restrict__ Xt,    // B,256,N f16 (transposed)
    const _Float16* __restrict__ Adj16, // B,N,N   f16
    const _Float16* __restrict__ W0h,   // 256,256 f16
    const float*    __restrict__ B0,    // 256
    const _Float16* __restrict__ W1h,   // 256,256 f16
    const float*    __restrict__ B1,    // 256
    const float*    __restrict__ Res,   // B,N,256 or null
    float*          __restrict__ Y,     // B,N,256 f32
    _Float16*       __restrict__ Yt)    // B,256,N f16
{
    __shared__ _Float16 sXrow[32 * FP];   // 32 x 256 row tile, stride 264
    __shared__ _Float16 sAgg[32 * FP];    // 32 x 256 agg tile, stride 264

    const int tid  = threadIdx.x;
    const int lane = tid & 31;
    const int wave = tid >> 5;
    const int b    = blockIdx.y;
    const int m0   = blockIdx.x * 32;

    // Stage the 32 x 256 X row tile (f32 -> f16, once per workgroup)
    for (int i = tid; i < 32 * FDIM; i += 128) {
        int r = i >> 8, c = i & 255;
        sXrow[r * FP + c] = (_Float16)X[((size_t)b * NPTS + m0 + r) * FDIM + c];
    }

    const int m     = lane & 15;        // A row / B col / C col in 16-tile
    const int kh8   = (lane >> 4) * 8;  // A K-half (ISA 7.12.2)
    const int kh16  = (lane >> 4) * 16; // unused; B K-half is kh8, same order as A
    const int kh2   = lane >> 4;        // C/D M-half
    const int fbase = wave * 64;

    // ------------------ Phase 1: agg = adj[m0:m0+32,:] @ X ------------------
    v8f acc[2][4];
    #pragma unroll
    for (int t = 0; t < 2; ++t)
        #pragma unroll
        for (int s = 0; s < 4; ++s)
            #pragma unroll
            for (int i = 0; i < 8; ++i) acc[t][s][i] = 0.f;

    const _Float16* arow0 = Adj16 + ((size_t)b * NPTS + m0 + m) * NPTS;
    const _Float16* arow1 = arow0 + (size_t)16 * NPTS;
    const _Float16* xcol  = Xt + ((size_t)b * FDIM + fbase + m) * NPTS + kh8;

    for (int k0 = 0; k0 < NPTS; k0 += 32) {
        AB16 a0, a1, bb[4];
        a0.u[0] = *(const uint4*)(arow0 + k0 + kh8);
        a0.u[1] = *(const uint4*)(arow0 + k0 + 16 + kh8);
        a1.u[0] = *(const uint4*)(arow1 + k0 + kh8);
        a1.u[1] = *(const uint4*)(arow1 + k0 + 16 + kh8);
        #pragma unroll
        for (int s = 0; s < 4; ++s) {
            const _Float16* p = xcol + (size_t)(s * 16) * NPTS + k0;
            bb[s].u[0] = *(const uint4*)(p);
            bb[s].u[1] = *(const uint4*)(p + 16);
        }
        if (k0 + 256 < NPTS) {
            __builtin_prefetch(arow0 + k0 + 256, 0, 1);
            __builtin_prefetch(arow1 + k0 + 256, 0, 1);
        }
        #pragma unroll
        for (int s = 0; s < 4; ++s) {
            acc[0][s] = __builtin_amdgcn_wmma_f32_16x16x32_f16(
                false, a0.h, false, bb[s].h, (short)0, acc[0][s], false, false);
            acc[1][s] = __builtin_amdgcn_wmma_f32_16x16x32_f16(
                false, a1.h, false, bb[s].h, (short)0, acc[1][s], false, false);
        }
    }

    // Share agg tile (rounded to f16) for phase 2
    #pragma unroll
    for (int t = 0; t < 2; ++t)
        #pragma unroll
        for (int s = 0; s < 4; ++s) {
            const int col = fbase + s * 16 + m;
            #pragma unroll
            for (int r = 0; r < 8; ++r)
                sAgg[(16 * t + 8 * kh2 + r) * FP + col] = (_Float16)acc[t][s][r];
        }
    __syncthreads();

    // ------------- Phase 2: Y = Xrow@W0^T + agg@W1^T + bias (+res) ----------
    v8f acc2[2][4];
    #pragma unroll
    for (int t = 0; t < 2; ++t)
        #pragma unroll
        for (int s = 0; s < 4; ++s)
            #pragma unroll
            for (int i = 0; i < 8; ++i) acc2[t][s][i] = 0.f;

    for (int k0 = 0; k0 < FDIM; k0 += 32) {
        AB16 ax[2], ag[2], w0v[4], w1v[4];
        #pragma unroll
        for (int t = 0; t < 2; ++t) {
            const _Float16* p1 = sXrow + (m + 16 * t) * FP + k0;
            ax[t].u[0] = *(const uint4*)(p1 + kh8);
            ax[t].u[1] = *(const uint4*)(p1 + 16 + kh8);
            const _Float16* p2 = sAgg + (m + 16 * t) * FP + k0;
            ag[t].u[0] = *(const uint4*)(p2 + kh8);
            ag[t].u[1] = *(const uint4*)(p2 + 16 + kh8);
        }
        #pragma unroll
        for (int s = 0; s < 4; ++s) {
            const int n = fbase + s * 16 + m;
            const _Float16* q0 = W0h + (size_t)n * FDIM + k0 + kh8;
            w0v[s].u[0] = *(const uint4*)(q0);
            w0v[s].u[1] = *(const uint4*)(q0 + 16);
            const _Float16* q1 = W1h + (size_t)n * FDIM + k0 + kh8;
            w1v[s].u[0] = *(const uint4*)(q1);
            w1v[s].u[1] = *(const uint4*)(q1 + 16);
        }
        #pragma unroll
        for (int s = 0; s < 4; ++s) {
            #pragma unroll
            for (int t = 0; t < 2; ++t) {
                acc2[t][s] = __builtin_amdgcn_wmma_f32_16x16x32_f16(
                    false, ax[t].h, false, w0v[s].h, (short)0, acc2[t][s], false, false);
                acc2[t][s] = __builtin_amdgcn_wmma_f32_16x16x32_f16(
                    false, ag[t].h, false, w1v[s].h, (short)0, acc2[t][s], false, false);
            }
        }
    }

    // Epilogue: bias (+residual) (+relu); write f32 row-major + f16 transposed
    #pragma unroll
    for (int t = 0; t < 2; ++t)
        #pragma unroll
        for (int s = 0; s < 4; ++s) {
            const int col = fbase + s * 16 + m;
            const float bias = B0[col] + B1[col];
            float v[8];
            #pragma unroll
            for (int r = 0; r < 8; ++r) {
                const int row = m0 + 16 * t + 8 * kh2 + r;
                float tt = acc2[t][s][r] + bias;
                if (Res) tt += Res[((size_t)b * NPTS + row) * SOUT + col];
                tt = fmaxf(tt, 0.f);   // every layer in this network applies ReLU
                v[r] = tt;
                Y[((size_t)b * NPTS + row) * SOUT + col] = tt;
            }
            union { uint4 u; unsigned w[4]; } pkv;
            #pragma unroll
            for (int j = 0; j < 4; ++j) pkv.w[j] = pk2(v[2 * j], v[2 * j + 1]);
            *(uint4*)(Yt + ((size_t)b * SOUT + col) * NPTS + m0 + 16 * t + 8 * kh2) = pkv.u;
        }
}

// ---------------------------------------------------------------------------
// FC head: pred = out@fcW^T + fcb ; g = pred*mask ; poly = bp + g
// ---------------------------------------------------------------------------
__global__ __launch_bounds__(256) void fc_kernel(
    const float* __restrict__ out,   // B,N,256
    const float* __restrict__ fcW,   // 2,256
    const float* __restrict__ fcb,   // 2
    const float* __restrict__ bp,    // B,N,2
    const float* __restrict__ mask,  // B,N
    float* __restrict__ pred,        // B,N,2
    float* __restrict__ poly,        // B,N,2
    float* __restrict__ g)           // B,N,2
{
    const int idx = blockIdx.x * blockDim.x + threadIdx.x;   // over B*N
    if (idx >= 16 * NPTS) return;
    const float* o = out + (size_t)idx * SOUT;
    float p0 = fcb[0], p1 = fcb[1];
    for (int k = 0; k < SOUT; ++k) {
        float v = o[k];
        p0 += v * fcW[k];
        p1 += v * fcW[SOUT + k];
    }
    pred[idx * 2 + 0] = p0;
    pred[idx * 2 + 1] = p1;
    const float mk = mask[idx];
    const float g0 = p0 * mk, g1 = p1 * mk;
    g[idx * 2 + 0] = g0;
    g[idx * 2 + 1] = g1;
    poly[idx * 2 + 0] = bp[idx * 2 + 0] + g0;
    poly[idx * 2 + 1] = bp[idx * 2 + 1] + g1;
}

// ---------------------------------------------------------------------------
// Laplacian energy: lap2-lap1 == g - adj@g. One block per batch.
// ---------------------------------------------------------------------------
__global__ __launch_bounds__(1024) void lap_kernel(
    const float* __restrict__ Adj,   // B,N,N f32
    const float* __restrict__ g,     // B,N,2
    float* __restrict__ energy)      // B
{
    __shared__ float red[1024];
    const int b = blockIdx.x;
    const int n = threadIdx.x;
    const float* arow = Adj + ((size_t)b * NPTS + n) * NPTS;
    const float* gb = g + (size_t)b * NPTS * 2;
    float s0 = 0.f, s1 = 0.f;
    for (int j = 0; j < NPTS; ++j) {
        float a = arow[j];
        s0 += a * gb[2 * j + 0];
        s1 += a * gb[2 * j + 1];
    }
    const float dx = gb[2 * n + 0] - s0;
    const float dy = gb[2 * n + 1] - s1;
    red[n] = sqrtf(dx * dx + dy * dy + 1e-10f);
    __syncthreads();
    for (int s = 512; s > 0; s >>= 1) {
        if (n < s) red[n] += red[n + s];
        __syncthreads();
    }
    if (n == 0) energy[b] = red[0] * (1.f / (float)NPTS);
}

// ---------------------------------------------------------------------------
extern "C" void kernel_launch(void* const* d_in, const int* in_sizes, int n_in,
                              void* d_out, int out_size, void* d_ws, size_t ws_size,
                              hipStream_t stream)
{
    (void)in_sizes; (void)n_in; (void)out_size; (void)ws_size;

    const float* features = (const float*)d_in[0];
    const float* base_pt  = (const float*)d_in[1];
    const float* adj      = (const float*)d_in[2];
    const float* mask     = (const float*)d_in[3];
    const float* first_W  = (const float*)d_in[4];
    const float* first_b  = (const float*)d_in[5];
    const float* mid_W    = (const float*)d_in[6];
    const float* mid_b    = (const float*)d_in[7];
    const float* last_W   = (const float*)d_in[8];
    const float* last_b   = (const float*)d_in[9];
    const float* fc_W     = (const float*)d_in[10];
    const float* fc_b     = (const float*)d_in[11];

    // ---- workspace carve (f32 region, then f16 region) ----
    float* ws   = (float*)d_ws;
    float* xbuf = ws;                                    // 16*1024*256
    float* bufA = xbuf + (size_t)16 * NPTS * FDIM;       // 16*1024*256 x3
    float* bufB = bufA + (size_t)16 * NPTS * SOUT;
    float* bufC = bufB + (size_t)16 * NPTS * SOUT;
    float* gbuf = bufC + (size_t)16 * NPTS * SOUT;       // 16*1024*2

    _Float16* hbase = (_Float16*)(gbuf + (size_t)16 * NPTS * 2);
    _Float16* adj16 = hbase;                             // 16*1024*1024
    _Float16* xt    = adj16 + (size_t)16 * NPTS * NPTS;  // 16*256*1024
    _Float16* bufAt = xt + (size_t)16 * FDIM * NPTS;     // 16*256*1024 x3
    _Float16* bufBt = bufAt + (size_t)16 * SOUT * NPTS;
    _Float16* bufCt = bufBt + (size_t)16 * SOUT * NPTS;
    _Float16* w16f  = bufCt + (size_t)16 * SOUT * NPTS;  // 2*256*256
    _Float16* w16m  = w16f + (size_t)2 * SOUT * FDIM;    // 6*4*256*256
    _Float16* w16l  = w16m + (size_t)24 * SOUT * SOUT;   // 2*256*256

    float* outp   = (float*)d_out;
    float* energy = outp;                 // 16
    float* poly   = outp + 16;            // 32768
    float* pred   = poly + 16 * NPTS * 2; // 32768

    // ---- one-time conversions ----
    interp_kernel<<<dim3(NPTS, 16), 256, 0, stream>>>(features, base_pt, xbuf, xt);
    padw16_kernel<<<(2 * 256 * FDIM + 255) / 256, 256, 0, stream>>>(first_W, w16f);
    {
        int n = 16 * NPTS * NPTS;
        cvt_h_kernel<<<(n / 4 + 255) / 256, 256, 0, stream>>>(adj, adj16, n);
    }
    {
        int n = 24 * SOUT * SOUT;
        cvt_h_kernel<<<(n / 4 + 255) / 256, 256, 0, stream>>>(mid_W, w16m, n);
    }
    {
        int n = 2 * SOUT * SOUT;
        cvt_h_kernel<<<(n / 4 + 255) / 256, 256, 0, stream>>>(last_W, w16l, n);
    }

    const dim3 gg(NPTS / 32, 16);

    // first layer (K padded 130 -> 256)
    gconv_kernel<<<gg, 128, 0, stream>>>(
        xbuf, xt, adj16, w16f, first_b, w16f + (size_t)SOUT * FDIM, first_b + SOUT,
        nullptr, bufA, bufAt);

    float*    cur  = bufA;  _Float16* curT = bufAt;
    float*    t1   = bufB;  _Float16* t1T  = bufBt;
    float*    t2   = bufC;  _Float16* t2T  = bufCt;
    for (int i = 0; i < 6; ++i) {
        const _Float16* W  = w16m + (size_t)i * 4 * SOUT * SOUT;
        const float*    bb = mid_b + (size_t)i * 4 * SOUT;
        gconv_kernel<<<gg, 128, 0, stream>>>(
            cur, curT, adj16, W, bb, W + (size_t)SOUT * SOUT, bb + SOUT,
            nullptr, t1, t1T);
        gconv_kernel<<<gg, 128, 0, stream>>>(
            t1, t1T, adj16, W + (size_t)2 * SOUT * SOUT, bb + 2 * SOUT,
            W + (size_t)3 * SOUT * SOUT, bb + 3 * SOUT,
            cur, t2, t2T);
        float* tf = cur; cur = t2; t2 = tf;
        _Float16* th = curT; curT = t2T; t2T = th;
    }
    gconv_kernel<<<gg, 128, 0, stream>>>(
        cur, curT, adj16, w16l, last_b, w16l + (size_t)SOUT * SOUT, last_b + SOUT,
        nullptr, t1, t1T);

    fc_kernel<<<(16 * NPTS + 255) / 256, 256, 0, stream>>>(
        t1, fc_W, fc_b, base_pt, mask, pred, poly, gbuf);
    lap_kernel<<<16, 1024, 0, stream>>>(adj, gbuf, energy);
}
